// SRHGNLayer_33028298506732
// MI455X (gfx1250) — compile-verified, matrix operand order checked by reading, not run on hardware
//
#include <hip/hip_runtime.h>
#include <math.h>

// ---------------------------------------------------------------------------
// SRHGN layer for MI455X (gfx1250, wave32).
// GEMMs: fp32 in -> bf16 fragments staged through double-buffered LDS ->
//        v_wmma_f32_16x16x32_bf16 (f32 accumulate).
//        64x128 macro-tile / 256 threads (8 waves); wave = 16x64 strip
//        -> 4 WMMA per 32-wide K step, software-pipelined K loop
//        (global b128 loads for step k+1 in flight while computing step k).
// Edge phase: scatter-max / scatter-sum softmax + per-feature atomic adds.
// ---------------------------------------------------------------------------

#define D 256
#define LEAKY 0.2f

typedef __attribute__((ext_vector_type(16))) __bf16 v16bf;
typedef __attribute__((ext_vector_type(8)))  float  v8f;

union FragB16 {              // one WMMA 16-bit operand = 8 dwords
    v16bf    v;
    uint4    q[2];
    unsigned u[8];
};

// ---------------- small helpers ----------------

__device__ __forceinline__ void atomicMaxF(float* addr, float value) {
    // signed-int max for non-negative floats, unsigned-int min for negatives.
    // Correct for buffers initialized to -INF, mixed-sign values.
    if (value >= 0.0f)
        atomicMax((int*)addr, __float_as_int(value));
    else
        atomicMin((unsigned int*)addr, __float_as_uint(value));
}

__device__ __forceinline__ float warp_sum(float v) {
    #pragma unroll
    for (int off = 16; off > 0; off >>= 1) v += __shfl_xor(v, off, 32);
    return v;
}

__device__ __forceinline__ unsigned pack_bf16(float x, float y) {
    union { __bf16 h[2]; unsigned u; } p;
    p.h[0] = (__bf16)x;              // -> v_cvt_pk_bf16_f32
    p.h[1] = (__bf16)y;
    return p.u;
}

// ---------------- tiny kernels ----------------

// u[k] = sum_j W[j][k] * a[j]   (W row-major [D,D]); one block of D threads.
__global__ void compute_u_kernel(const float* __restrict__ W,
                                 const float* __restrict__ a,
                                 float* __restrict__ u) {
    int k = threadIdx.x;
    float s = 0.0f;
    for (int j = 0; j < D; ++j) s += W[(size_t)j * D + k] * a[j];
    u[k] = s;
}

__global__ void fill_kernel(float* __restrict__ p, float v, int n) {
    int i = blockIdx.x * blockDim.x + threadIdx.x;
    if (i < n) p[i] = v;
}

// s[row] = dot(X[row,:], u)  -- one wave per row, 8 rows per 256-thread block.
__global__ void rowdot_kernel(const float* __restrict__ X,
                              const float* __restrict__ u,
                              float* __restrict__ s, int N) {
    int row  = blockIdx.x * (blockDim.x >> 5) + (threadIdx.x >> 5);
    int lane = threadIdx.x & 31;
    if (row >= N) return;
    const float* xr = X + (size_t)row * D;
    float acc = 0.0f;
    #pragma unroll
    for (int i = 0; i < D / 32; ++i) {
        int k = lane + 32 * i;
        acc = fmaf(xr[k], u[k], acc);
    }
    acc = warp_sum(acc);
    if (lane == 0) s[row] = acc;
}

// ---------------- WMMA GEMM: C = A @ W^T (+bias), A:[M,D], W:[D,D] ----------
// Macro-tile 64(M) x 128(N) per 256-thread block (8 waves).
// Wave w: row-tile (w&3)*16, col-group (w>>2)*64 -> 4 x 16x16 WMMA tiles.
// LDS (double-buffered): bf16 pairs packed in dwords, [row][16 dwords] so a
// WMMA fragment (ISA 7.12.2: lane L -> M/N=L&15, half=L>>4; word v holds
// K pair k0(v) = (v<4 ? 2v : 16+2(v-4)) + 8*half) is exactly 2 x ds_load_b128:
// dwords [4*half .. 4*half+3] and [8+4*half .. 8+4*half+3].
__global__ void __launch_bounds__(256)
wmma_gemm_bf16(const float* __restrict__ A,
               const float* __restrict__ W,
               const float* __restrict__ bias,
               float* __restrict__ C, int M) {
    __shared__ __align__(16) unsigned ldsA[2][64  * 16];   // 2 x 4 KB
    __shared__ __align__(16) unsigned ldsB[2][128 * 16];   // 2 x 8 KB

    const int tileM = blockIdx.x * 64;
    const int tileN = blockIdx.y * 128;

    const int tid   = threadIdx.x;
    const int lane  = tid & 31;
    const int wv    = tid >> 5;          // wave 0..7
    const int wm    = wv & 3;            // row tile within macro-tile
    const int wn    = wv >> 2;           // col group (0..1), 64 cols each
    const int half  = lane >> 4;
    const int idx16 = lane & 15;

    v8f acc[4] = {{}, {}, {}, {}};
    float4 ra[2], rb[4];                 // in-flight staging registers

    // issue all global b128 loads for K chunk `k` into ra/rb (no waits between)
    auto gload = [&](int k) {
        #pragma unroll
        for (int i = 0; i < 2; ++i) {    // A: 64 rows x 8 float4 = 512 jobs
            int j  = tid + 256 * i;
            int r  = j >> 3;             // 0..63
            int kk = (j & 7) << 2;       // 0,4,..,28
            int grow = tileM + r;
            if (grow >= M) grow = M - 1; // clamp (stores are guarded)
            ra[i] = *(const float4*)(A + (size_t)grow * D + k + kk);
        }
        #pragma unroll
        for (int i = 0; i < 4; ++i) {    // B[k][n] = W[n][k]: 128 rows x 8 f4
            int j  = tid + 256 * i;
            int n  = j >> 3;             // 0..127
            int kk = (j & 7) << 2;
            rb[i] = *(const float4*)(W + (size_t)(tileN + n) * D + k + kk);
        }
        if (k + 32 < D) {                // keep a far prefetch in flight
            int grow = tileM + (wm << 4) + idx16;
            if (grow >= M) grow = M - 1;
            __builtin_prefetch(A + (size_t)grow * D + k + 32, 0, 1);
        }
    };

    // convert + store staged registers into LDS buffer `buf` (ds_store_b64)
    auto sstore = [&](int buf) {
        #pragma unroll
        for (int i = 0; i < 2; ++i) {
            int j  = tid + 256 * i;
            int r  = j >> 3;
            int kw = (j & 7) << 1;       // dword index within row
            uint2 p;
            p.x = pack_bf16(ra[i].x, ra[i].y);
            p.y = pack_bf16(ra[i].z, ra[i].w);
            *(uint2*)&ldsA[buf][(r << 4) + kw] = p;
        }
        #pragma unroll
        for (int i = 0; i < 4; ++i) {
            int j  = tid + 256 * i;
            int n  = j >> 3;
            int kw = (j & 7) << 1;
            uint2 p;
            p.x = pack_bf16(rb[i].x, rb[i].y);
            p.y = pack_bf16(rb[i].z, rb[i].w);
            *(uint2*)&ldsB[buf][(n << 4) + kw] = p;
        }
    };

    // fragments from LDS buffer `buf` + 4 WMMA
    auto comp = [&](int buf) {
        FragB16 a;
        const unsigned* baseA =
            &ldsA[buf][(((wm << 4) + idx16) << 4) + (half << 2)];
        a.q[0] = *(const uint4*)(baseA);
        a.q[1] = *(const uint4*)(baseA + 8);
        #pragma unroll
        for (int c = 0; c < 4; ++c) {
            FragB16 b;
            const unsigned* baseB =
                &ldsB[buf][(((wn << 6) + (c << 4) + idx16) << 4) + (half << 2)];
            b.q[0] = *(const uint4*)(baseB);
            b.q[1] = *(const uint4*)(baseB + 8);
            acc[c] = __builtin_amdgcn_wmma_f32_16x16x32_bf16(
                         false, a.v, false, b.v, (short)0, acc[c], false, false);
        }
    };

    // ---- software pipeline over 8 K-steps ----
    gload(0);
    sstore(0);
    __syncthreads();
    #pragma unroll
    for (int ks = 0; ks < D / 32; ++ks) {
        if (ks + 1 < D / 32) gload((ks + 1) * 32);  // overlap with compute
        comp(ks & 1);
        if (ks + 1 < D / 32) {
            sstore((ks + 1) & 1);                   // other buffer: no WAR hazard
            __syncthreads();                        // one barrier per step
        }
    }

    // ---- store: C layout dword r -> row r + 8*half, col = idx16 ----
    #pragma unroll
    for (int c = 0; c < 4; ++c) {
        const int col  = tileN + (wn << 6) + (c << 4) + idx16;
        const float badd = bias ? bias[col] : 0.0f;
        #pragma unroll
        for (int r = 0; r < 8; ++r) {
            const int row = tileM + (wm << 4) + r + 8 * half;
            if (row < M) C[(size_t)row * D + col] = acc[c][r] + badd;
        }
    }
}

// ---------------- edge phase ----------------

__global__ void edge_score_max_kernel(const int* __restrict__ row,
                                      const int* __restrict__ col,
                                      const float* __restrict__ ssrc,
                                      const float* __restrict__ sdst,
                                      float* __restrict__ score,
                                      float* __restrict__ m, int E) {
    int e = blockIdx.x * blockDim.x + threadIdx.x;
    if (e >= E) return;
    float s = ssrc[row[e]] + sdst[col[e]];
    s = (s >= 0.0f) ? s : LEAKY * s;          // leaky_relu
    score[e] = s;
    atomicMaxF(&m[col[e]], s);
}

__global__ void edge_exp_sum_kernel(const int* __restrict__ col,
                                    const float* __restrict__ score,
                                    const float* __restrict__ m,
                                    float* __restrict__ eval,
                                    float* __restrict__ denom, int E) {
    int e = blockIdx.x * blockDim.x + threadIdx.x;
    if (e >= E) return;
    int c = col[e];
    float mm = m[c];
    if (isinf(mm)) mm = 0.0f;                 // match reference clamp
    float ev = __expf(score[e] - mm);
    eval[e] = ev;
    atomicAdd(&denom[c], ev);
}

// One wave per edge; lane covers 8 features. agg[c,:] += attn * wx[r,:]
__global__ void edge_aggregate_kernel(const int* __restrict__ row,
                                      const int* __restrict__ col,
                                      const float* __restrict__ wx,
                                      const float* __restrict__ eval,
                                      const float* __restrict__ denom,
                                      float* __restrict__ agg, int E) {
    int e    = blockIdx.x * (blockDim.x >> 5) + (threadIdx.x >> 5);
    int lane = threadIdx.x & 31;
    if (e >= E) return;
    const int r = row[e], c = col[e];
    float d = denom[c];
    if (d == 0.0f) d = 1.0f;
    const float attn = eval[e] / d;
    const float* src = wx  + (size_t)r * D;
    float*       dst = agg + (size_t)c * D;
    #pragma unroll
    for (int i = 0; i < D / 32; ++i) {
        int f = lane + 32 * i;
        atomicAdd(&dst[f], src[f] * attn);    // global_atomic_add_f32
    }
}

// ---------------- fusion ----------------

__device__ __forceinline__ float elu_f(float v) {
    return (v > 0.0f) ? v : (expm1f(v));
}

// Single relation: softmax over R=1 is identity -> out = elu(self + agg).
__global__ void fuse_single_kernel(const float* __restrict__ agg,
                                   float* __restrict__ out, int n) {
    int i = blockIdx.x * blockDim.x + threadIdx.x;
    if (i < n) out[i] = elu_f(out[i] + agg[i]);
}

// Two relations: type attention with tanh(agg).q scores, then elu.
__global__ void fuse_pair_kernel(const float* __restrict__ agg0,
                                 const float* __restrict__ agg1,
                                 const float* __restrict__ q,
                                 float* __restrict__ out, int N) {
    int rowi = blockIdx.x * (blockDim.x >> 5) + (threadIdx.x >> 5);
    int lane = threadIdx.x & 31;
    if (rowi >= N) return;
    const float* a0 = agg0 + (size_t)rowi * D;
    const float* a1 = agg1 + (size_t)rowi * D;
    float*       o  = out  + (size_t)rowi * D;
    float s0 = 0.0f, s1 = 0.0f;
    #pragma unroll
    for (int i = 0; i < D / 32; ++i) {
        int k = lane + 32 * i;
        float qv = q[k];
        s0 = fmaf(tanhf(a0[k]), qv, s0);
        s1 = fmaf(tanhf(a1[k]), qv, s1);
    }
    s0 = warp_sum(s0);
    s1 = warp_sum(s1);
    float mx = fmaxf(s0, s1);
    float e0 = __expf(s0 - mx), e1 = __expf(s1 - mx);
    float inv = 1.0f / (e0 + e1);
    float w0 = e0 * inv, w1 = e1 * inv;
    #pragma unroll
    for (int i = 0; i < D / 32; ++i) {
        int k = lane + 32 * i;
        o[k] = elu_f(o[k] + w0 * a0[k] + w1 * a1[k]);
    }
}

// ---------------- host orchestration ----------------

static inline dim3 gemm_grid(int M) { return dim3((M + 63) / 64, D / 128, 1); }

struct RelPlan {
    const float* x_src; int Ns;
    const float* x_dst; int Nd;
    const int*   row;   const int* col; int E;
    const float* W; const float* a_src; const float* a_dst;
    float* wx; float* agg;            // [Ns*D], [Nd*D]
    float* s_src; float* s_dst;       // [Ns], [Nd]
    float* m; float* denom;           // [Nd]
};

static void run_relation(const RelPlan& p, float* u_src, float* u_dst,
                         float* score, float* eval, hipStream_t stream) {
    // attention projection vectors (tiny)
    compute_u_kernel<<<1, D, 0, stream>>>(p.W, p.a_src, u_src);
    compute_u_kernel<<<1, D, 0, stream>>>(p.W, p.a_dst, u_dst);
    // wx_src via WMMA GEMM
    wmma_gemm_bf16<<<gemm_grid(p.Ns), 256, 0, stream>>>(p.x_src, p.W, nullptr, p.wx, p.Ns);
    // per-node scores
    rowdot_kernel<<<(p.Ns + 7) / 8, 256, 0, stream>>>(p.x_src, u_src, p.s_src, p.Ns);
    rowdot_kernel<<<(p.Nd + 7) / 8, 256, 0, stream>>>(p.x_dst, u_dst, p.s_dst, p.Nd);
    // init reduction buffers
    fill_kernel<<<(p.Nd + 255) / 256, 256, 0, stream>>>(p.m, -INFINITY, p.Nd);
    fill_kernel<<<(p.Nd + 255) / 256, 256, 0, stream>>>(p.denom, 0.0f, p.Nd);
    {
        int n = p.Nd * D;
        fill_kernel<<<(n + 255) / 256, 256, 0, stream>>>(p.agg, 0.0f, n);
    }
    // scatter softmax + aggregate
    edge_score_max_kernel<<<(p.E + 255) / 256, 256, 0, stream>>>(
        p.row, p.col, p.s_src, p.s_dst, score, p.m, p.E);
    edge_exp_sum_kernel<<<(p.E + 255) / 256, 256, 0, stream>>>(
        p.col, score, p.m, eval, p.denom, p.E);
    edge_aggregate_kernel<<<(p.E + 7) / 8, 256, 0, stream>>>(
        p.row, p.col, p.wx, eval, p.denom, p.agg, p.E);
}

extern "C" void kernel_launch(void* const* d_in, const int* in_sizes, int n_in,
                              void* d_out, int out_size, void* d_ws, size_t ws_size,
                              hipStream_t stream) {
    (void)n_in; (void)out_size; (void)ws_size;

    const float* h_author = (const float*)d_in[0];
    const float* h_paper  = (const float*)d_in[1];
    const int*   ei_ap    = (const int*)d_in[2];
    const int*   ei_pa    = (const int*)d_in[3];
    const int*   ei_pp    = (const int*)d_in[4];
    const float* W_ap = (const float*)d_in[5];
    const float* a_src_ap = (const float*)d_in[6];
    const float* a_dst_ap = (const float*)d_in[7];
    const float* W_pa = (const float*)d_in[8];
    const float* a_src_pa = (const float*)d_in[9];
    const float* a_dst_pa = (const float*)d_in[10];
    const float* W_pp = (const float*)d_in[11];
    const float* a_src_pp = (const float*)d_in[12];
    const float* a_dst_pp = (const float*)d_in[13];
    const float* Wself_a = (const float*)d_in[14];
    const float* bself_a = (const float*)d_in[15];
    const float* Wself_p = (const float*)d_in[16];
    const float* bself_p = (const float*)d_in[17];
    const float* q_paper = (const float*)d_in[19];
    // q_author (d_in[18]) unused: softmax over a single relation is identity.

    const int Na = in_sizes[0] / D;
    const int Np = in_sizes[1] / D;
    const int E_ap = in_sizes[2] / 2;
    const int E_pa = in_sizes[3] / 2;
    const int E_pp = in_sizes[4] / 2;
    int Emax = E_ap > E_pa ? E_ap : E_pa;
    if (E_pp > Emax) Emax = E_pp;

    // workspace carve-up (floats)
    float* ws = (float*)d_ws;
    size_t off = 0;
    auto alloc = [&](size_t n) { float* p = ws + off; off += n; return p; };

    float* wx_ap  = alloc((size_t)Na * D);
    float* wx_pa  = alloc((size_t)Np * D);
    float* wx_pp  = alloc((size_t)Np * D);
    float* agg_ap = alloc((size_t)Np * D);
    float* agg_pa = alloc((size_t)Na * D);
    float* agg_pp = alloc((size_t)Np * D);
    float* s_bufs = alloc((size_t)2 * (Na + Np) + 2 * (size_t)Np); // 6 score vecs
    float* m_ap = alloc(Np); float* dn_ap = alloc(Np);
    float* m_pa = alloc(Na); float* dn_pa = alloc(Na);
    float* m_pp = alloc(Np); float* dn_pp = alloc(Np);
    float* score = alloc(Emax);
    float* eval  = alloc(Emax);
    float* u_src = alloc(D);
    float* u_dst = alloc(D);

    float* ss_ap = s_bufs;           // [Na]
    float* sd_ap = ss_ap + Na;       // [Np]
    float* ss_pa = sd_ap + Np;       // [Np]
    float* sd_pa = ss_pa + Np;       // [Na]
    float* ss_pp = sd_pa + Na;       // [Np]
    float* sd_pp = ss_pp + Np;       // [Np]

    // --- relation ap: author -> paper ---
    RelPlan ap = { h_author, Na, h_paper, Np, ei_ap, ei_ap + E_ap, E_ap,
                   W_ap, a_src_ap, a_dst_ap, wx_ap, agg_ap, ss_ap, sd_ap, m_ap, dn_ap };
    run_relation(ap, u_src, u_dst, score, eval, stream);
    // --- relation pa: paper -> author ---
    RelPlan pa = { h_paper, Np, h_author, Na, ei_pa, ei_pa + E_pa, E_pa,
                   W_pa, a_src_pa, a_dst_pa, wx_pa, agg_pa, ss_pa, sd_pa, m_pa, dn_pa };
    run_relation(pa, u_src, u_dst, score, eval, stream);
    // --- relation pp: paper -> paper ---
    RelPlan pp = { h_paper, Np, h_paper, Np, ei_pp, ei_pp + E_pp, E_pp,
                   W_pp, a_src_pp, a_dst_pp, wx_pp, agg_pp, ss_pp, sd_pp, m_pp, dn_pp };
    run_relation(pp, u_src, u_dst, score, eval, stream);

    // --- self transforms straight into d_out (author block, then paper) ---
    float* out_a = (float*)d_out;
    float* out_p = out_a + (size_t)Na * D;
    wmma_gemm_bf16<<<gemm_grid(Na), 256, 0, stream>>>(h_author, Wself_a, bself_a, out_a, Na);
    wmma_gemm_bf16<<<gemm_grid(Np), 256, 0, stream>>>(h_paper,  Wself_p, bself_p, out_p, Np);

    // --- fusion + elu (in place on d_out) ---
    {
        int n = Na * D;
        fuse_single_kernel<<<(n + 255) / 256, 256, 0, stream>>>(agg_pa, out_a, n);
    }
    fuse_pair_kernel<<<(Np + 7) / 8, 256, 0, stream>>>(agg_ap, agg_pp, q_paper, out_p, Np);
}
